// PointNet2SAMSG_27608049778807
// MI455X (gfx1250) — compile-verified, matrix operand order checked by reading, not run on hardware
//
#include <hip/hip_runtime.h>

typedef __attribute__((ext_vector_type(16))) _Float16 v16h;
typedef __attribute__((ext_vector_type(8)))  _Float16 v8h;
typedef __attribute__((ext_vector_type(8)))  float    v8f;

constexpr int BB     = 4;
constexpr int NN     = 8192;
constexpr int KK     = 2048;
constexpr int CIN_F  = 64;
constexpr int CAT_CH = 512;
constexpr int WPAD   = 264;   // halves per LDS row (16B-aligned rows, bank-skewed)

// ---------------------------------------------------------------------------
// Farthest point sampling: one block per batch. min-distance state lives in
// registers (8 points/thread), per-iteration argmax via wave shuffles + LDS.
// ---------------------------------------------------------------------------
__global__ __launch_bounds__(1024) void fps_kernel(const float* __restrict__ points,
                                                   int* __restrict__ fps_idx,
                                                   float* __restrict__ sampled) {
  const int b    = blockIdx.x;
  const int tid  = threadIdx.x;
  const int lane = tid & 31;
  const int wid  = tid >> 5;
  const float* pb = points + (size_t)b * NN * 3;

  float px[8], py[8], pz[8], mind[8];
  int   nn[8];
#pragma unroll
  for (int i = 0; i < 8; ++i) {
    int n = i * 1024 + tid;
    nn[i] = n;
    px[i] = pb[n * 3 + 0];
    py[i] = pb[n * 3 + 1];
    pz[i] = pb[n * 3 + 2];
    mind[i] = 1e10f;
  }

  __shared__ float s_val[32];
  __shared__ int   s_idx[32];
  __shared__ int   s_last;

  if (tid == 0) {
    fps_idx[(size_t)b * KK] = 0;
    sampled[((size_t)b * KK) * 3 + 0] = pb[0];
    sampled[((size_t)b * KK) * 3 + 1] = pb[1];
    sampled[((size_t)b * KK) * 3 + 2] = pb[2];
    s_last = 0;
  }
  __syncthreads();

  int last = 0;
  for (int it = 1; it < KK; ++it) {
    const float cx = pb[last * 3 + 0];
    const float cy = pb[last * 3 + 1];
    const float cz = pb[last * 3 + 2];
    float bv = -1.f; int bn = 0;
#pragma unroll
    for (int i = 0; i < 8; ++i) {
      float dx = px[i] - cx, dy = py[i] - cy, dz = pz[i] - cz;
      float d = dx * dx + dy * dy + dz * dz;
      mind[i] = fminf(mind[i], d);
      if (mind[i] > bv) { bv = mind[i]; bn = nn[i]; }
    }
#pragma unroll
    for (int off = 16; off >= 1; off >>= 1) {
      float ov = __shfl_xor(bv, off, 32);
      int   on = __shfl_xor(bn, off, 32);
      if (ov > bv || (ov == bv && on < bn)) { bv = ov; bn = on; }
    }
    if (lane == 0) { s_val[wid] = bv; s_idx[wid] = bn; }
    __syncthreads();
    if (wid == 0) {
      float v = s_val[lane]; int n2 = s_idx[lane];
#pragma unroll
      for (int off = 16; off >= 1; off >>= 1) {
        float ov = __shfl_xor(v, off, 32);
        int   on = __shfl_xor(n2, off, 32);
        if (ov > v || (ov == v && on < n2)) { v = ov; n2 = on; }
      }
      if (lane == 0) {
        s_last = n2;
        fps_idx[(size_t)b * KK + it] = n2;
        sampled[((size_t)b * KK + it) * 3 + 0] = pb[n2 * 3 + 0];
        sampled[((size_t)b * KK + it) * 3 + 1] = pb[n2 * 3 + 1];
        sampled[((size_t)b * KK + it) * 3 + 2] = pb[n2 * 3 + 2];
      }
    }
    __syncthreads();
    last = s_last;
  }
}

// ---------------------------------------------------------------------------
// Ball query for all 3 radii in one scan: one wave per (b,k) centroid.
// ---------------------------------------------------------------------------
__global__ __launch_bounds__(128) void ballq_kernel(const float* __restrict__ points,
                                                    const float* __restrict__ sampled,
                                                    int* __restrict__ idx1,
                                                    int* __restrict__ idx2,
                                                    int* __restrict__ idx3) {
  const int q    = blockIdx.x * 4 + (threadIdx.x >> 5);
  const int lane = threadIdx.x & 31;
  const int b    = q / KK;
  const float qx = sampled[(size_t)q * 3 + 0];
  const float qy = sampled[(size_t)q * 3 + 1];
  const float qz = sampled[(size_t)q * 3 + 2];
  const float* pb = points + (size_t)b * NN * 3;
  const float r1 = 0.2f * 0.2f, r2 = 0.4f * 0.4f, r3 = 0.8f * 0.8f;
  const unsigned below = (1u << lane) - 1u;

  int c1 = 0, c2 = 0, c3 = 0, f1 = 0, f2 = 0, f3 = 0;
  for (int n0 = 0; n0 < NN; n0 += 32) {
    const int n = n0 + lane;
    float dx = pb[n * 3 + 0] - qx;
    float dy = pb[n * 3 + 1] - qy;
    float dz = pb[n * 3 + 2] - qz;
    float d2 = dx * dx + dy * dy + dz * dz;
    unsigned m1 = (unsigned)__ballot(d2 < r1);
    unsigned m2 = (unsigned)__ballot(d2 < r2);
    unsigned m3 = (unsigned)__ballot(d2 < r3);
    if (c1 < 16 && m1) {
      if (c1 == 0) f1 = n0 + __builtin_ctz(m1);
      int rank = c1 + __popc(m1 & below);
      if ((d2 < r1) && rank < 16) idx1[(size_t)q * 16 + rank] = n;
      c1 += __popc(m1);
    }
    if (c2 < 32 && m2) {
      if (c2 == 0) f2 = n0 + __builtin_ctz(m2);
      int rank = c2 + __popc(m2 & below);
      if ((d2 < r2) && rank < 32) idx2[(size_t)q * 32 + rank] = n;
      c2 += __popc(m2);
    }
    if (c3 < 64 && m3) {
      if (c3 == 0) f3 = n0 + __builtin_ctz(m3);
      int rank = c3 + __popc(m3 & below);
      if ((d2 < r3) && rank < 64) idx3[(size_t)q * 64 + rank] = n;
      c3 += __popc(m3);
    }
    if (c1 >= 16 && c2 >= 32 && c3 >= 64) break;
  }
  const int fill1 = c1 ? f1 : 0;
  const int fill2 = c2 ? f2 : 0;
  const int fill3 = c3 ? f3 : 0;
  for (int pos = lane; pos < 16; pos += 32) if (pos >= c1) idx1[(size_t)q * 16 + pos] = fill1;
  for (int pos = lane; pos < 32; pos += 32) if (pos >= c2) idx2[(size_t)q * 32 + pos] = fill2;
  for (int pos = lane; pos < 64; pos += 32) if (pos >= c3) idx3[(size_t)q * 64 + pos] = fill3;
}

// ---------------------------------------------------------------------------
// Weight pre-pack: f32 [COUT, CINR] -> f16 WMMA B-fragment layout.
// ---------------------------------------------------------------------------
__global__ __launch_bounds__(256) void pack_w_kernel(const float* __restrict__ w,
                                                     _Float16* __restrict__ out,
                                                     int COUT, int CINR, int CINP) {
  const int idx = blockIdx.x * 256 + threadIdx.x;
  const int total = COUT * CINP;
  if (idx >= total) return;
  const int h    = idx & 15;
  const int lane = (idx >> 4) & 31;
  const int frag = idx >> 9;            // nt*KC + kc
  const int KC = CINP / 32;
  const int kc = frag % KC;
  const int nt = frag / KC;
  const int o  = nt * 16 + (lane & 15);
  const int k0 = kc * 32 + (lane >> 4) * 8;
  const int k  = (h < 8) ? (k0 + h) : (k0 + 16 + (h - 8));
  const float v = (k < CINR) ? w[(size_t)o * CINR + k] : 0.f;
  out[idx] = (_Float16)v;
}

// ---------------------------------------------------------------------------
// One MLP layer on a wave's 32-row (2 M-block) strip. Each B fragment now
// feeds TWO WMMAs (halves weight traffic / load stalls vs one M-block).
// N-dim processed in passes of <=8 tiles to cap accumulators at 128 VGPRs.
// A fragments: ds_load_b128 pairs from the wave's LDS slab (re-read per pass,
// LDS-cheap). B fragments: coalesced global_load_b128 pairs of packed f16.
// ---------------------------------------------------------------------------
template<int CINP, int COUT>
__device__ __forceinline__ void mlp_layer(const _Float16 (&in)[64][WPAD],
                                          _Float16 (&out)[64][WPAD],
                                          int rowBase, int lane,
                                          const _Float16* __restrict__ wp,
                                          const float* __restrict__ sc,
                                          const float* __restrict__ sh) {
  constexpr int KC  = CINP / 32;
  constexpr int NT  = COUT / 16;
  constexpr int NTC = (NT > 8) ? 8 : NT;   // N tiles per pass
  constexpr int NP  = NT / NTC;            // passes
  const int rsel = lane & 15;
  const int hsel = lane >> 4;
  const v8f zero = {0.f, 0.f, 0.f, 0.f, 0.f, 0.f, 0.f, 0.f};

  for (int np = 0; np < NP; ++np) {
    v8f acc[2][NTC];
#pragma unroll
    for (int nt = 0; nt < NTC; ++nt) { acc[0][nt] = zero; acc[1][nt] = zero; }

    for (int kc = 0; kc < KC; ++kc) {
      const int k0 = kc * 32 + hsel * 8;
      const _Float16* rp0 = &in[rowBase + rsel][0];
      const _Float16* rp1 = &in[rowBase + 16 + rsel][0];
      v8h a0lo = *(const v8h*)(rp0 + k0);
      v8h a0hi = *(const v8h*)(rp0 + k0 + 16);
      v8h a1lo = *(const v8h*)(rp1 + k0);
      v8h a1hi = *(const v8h*)(rp1 + k0 + 16);
      v16h a0, a1;
#pragma unroll
      for (int i = 0; i < 8; ++i) {
        a0[i] = a0lo[i]; a0[i + 8] = a0hi[i];
        a1[i] = a1lo[i]; a1[i + 8] = a1hi[i];
      }
#pragma unroll
      for (int nt = 0; nt < NTC; ++nt) {
        const int ntg = np * NTC + nt;
        const v8h* bp = (const v8h*)(wp + (((size_t)(ntg * KC + kc) * 32 + lane) << 4));
        v8h blo = bp[0];
        v8h bhi = bp[1];
        v16h bf;
#pragma unroll
        for (int i = 0; i < 8; ++i) { bf[i] = blo[i]; bf[i + 8] = bhi[i]; }
        acc[0][nt] = __builtin_amdgcn_wmma_f32_16x16x32_f16(false, a0, false, bf, (short)0,
                                                            acc[0][nt], false, false);
        acc[1][nt] = __builtin_amdgcn_wmma_f32_16x16x32_f16(false, a1, false, bf, (short)0,
                                                            acc[1][nt], false, false);
      }
    }
#pragma unroll
    for (int nt = 0; nt < NTC; ++nt) {
      const int o = (np * NTC + nt) * 16 + rsel;
      const float s = sc[o], h = sh[o];
#pragma unroll
      for (int mb = 0; mb < 2; ++mb) {
#pragma unroll
        for (int r = 0; r < 8; ++r) {
          const int m = mb * 16 + r + hsel * 8;   // C/D: VGPR r, lane>=16 -> M+8
          float v = acc[mb][nt][r] * s + h;
          v = v > 0.f ? v : 0.f;
          out[rowBase + m][o] = (_Float16)v;
        }
      }
    }
  }
}

// ---------------------------------------------------------------------------
// Fused per-scale kernel: 64 threads = 2 waves x 32 rows = 64 points/block.
// Gather grouped input into LDS as f16, run 3 WMMA layers (LDS ping-pong;
// same-wave DS ops in-order), then max-pool over S into cat[B*K,512].
// ---------------------------------------------------------------------------
template<int S, int C1, int C2, int C3, int CATOFF>
__global__ __launch_bounds__(64) void sa_mlp_kernel(
    const float* __restrict__ points, const float* __restrict__ features,
    const float* __restrict__ sampled, const int* __restrict__ idx,
    const _Float16* __restrict__ pw1, const float* __restrict__ sc1, const float* __restrict__ sh1,
    const _Float16* __restrict__ pw2, const float* __restrict__ sc2, const float* __restrict__ sh2,
    const _Float16* __restrict__ pw3, const float* __restrict__ sc3, const float* __restrict__ sh3,
    _Float16* __restrict__ cat) {
  __shared__ __align__(16) _Float16 Xa[64][WPAD];
  __shared__ __align__(16) _Float16 Xb[64][WPAD];
  const int wid     = threadIdx.x >> 5;
  const int lane    = threadIdx.x & 31;
  const int rowBase = wid * 32;

  {  // gather: thread t fills row t (96 padded channels)
    const int t = threadIdx.x;
    const int p = blockIdx.x * 64 + t;
    const int q = p / S;               // global centroid id = b*KK + k
    const int bq = q / KK;
    const int j = idx[(size_t)p];
#pragma unroll 4
    for (int c = 0; c < 96; ++c) {
      float v;
      if (c < 3)       v = points[((size_t)bq * NN + j) * 3 + c] - sampled[(size_t)q * 3 + c];
      else if (c < 67) v = features[((size_t)bq * CIN_F + (c - 3)) * NN + j];
      else             v = 0.f;
      Xa[t][c] = (_Float16)v;
    }
  }

  mlp_layer<96, C1>(Xa, Xb, rowBase, lane, pw1, sc1, sh1);
  mlp_layer<C1, C2>(Xb, Xa, rowBase, lane, pw2, sc2, sh2);
  mlp_layer<C2, C3>(Xa, Xb, rowBase, lane, pw3, sc3, sh3);
  __syncthreads();

  constexpr int CT = 64 / S;           // centroids per block
  for (int t = threadIdx.x; t < CT * C3; t += 64) {
    const int ci = t / C3, o = t - ci * C3;
    float mv = -3.0e38f;
    for (int s = 0; s < S; ++s) {
      float v = (float)Xb[ci * S + s][o];
      mv = v > mv ? v : mv;
    }
    const int gq = (blockIdx.x * 64) / S + ci;
    cat[(size_t)gq * CAT_CH + CATOFF + o] = (_Float16)mv;
  }
}

// ---------------------------------------------------------------------------
// Aggregation: [B*K,512] f16 x packed f16 weights -> relu -> [B,256,K].
// 64 threads = 2 waves x 32 rows; 2 N-passes of 8 tiles; MB=2 per wave.
// ---------------------------------------------------------------------------
__global__ __launch_bounds__(64) void agg_kernel(const _Float16* __restrict__ cat,
                                                 const _Float16* __restrict__ pw,
                                                 const float* __restrict__ sc,
                                                 const float* __restrict__ sh,
                                                 float* __restrict__ out) {
  const int wid     = threadIdx.x >> 5;
  const int lane    = threadIdx.x & 31;
  const int rowBase = blockIdx.x * 64 + wid * 32;
  const int rsel = lane & 15, hsel = lane >> 4;
  constexpr int KC = CAT_CH / 32;      // 16
  const v8f zero = {0.f, 0.f, 0.f, 0.f, 0.f, 0.f, 0.f, 0.f};

  const _Float16* arow0 = cat + (size_t)(rowBase + rsel) * CAT_CH;
  const _Float16* arow1 = cat + (size_t)(rowBase + 16 + rsel) * CAT_CH;

  for (int np = 0; np < 2; ++np) {
    v8f acc[2][8];
#pragma unroll
    for (int nt = 0; nt < 8; ++nt) { acc[0][nt] = zero; acc[1][nt] = zero; }

    for (int kc = 0; kc < KC; ++kc) {
      const int k0 = kc * 32 + hsel * 8;
      v8h a0lo = *(const v8h*)(arow0 + k0);
      v8h a0hi = *(const v8h*)(arow0 + k0 + 16);
      v8h a1lo = *(const v8h*)(arow1 + k0);
      v8h a1hi = *(const v8h*)(arow1 + k0 + 16);
      v16h a0, a1;
#pragma unroll
      for (int i = 0; i < 8; ++i) {
        a0[i] = a0lo[i]; a0[i + 8] = a0hi[i];
        a1[i] = a1lo[i]; a1[i + 8] = a1hi[i];
      }
#pragma unroll
      for (int nt = 0; nt < 8; ++nt) {
        const int ntg = np * 8 + nt;
        const v8h* bp = (const v8h*)(pw + (((size_t)(ntg * KC + kc) * 32 + lane) << 4));
        v8h blo = bp[0];
        v8h bhi = bp[1];
        v16h bf;
#pragma unroll
        for (int i = 0; i < 8; ++i) { bf[i] = blo[i]; bf[i + 8] = bhi[i]; }
        acc[0][nt] = __builtin_amdgcn_wmma_f32_16x16x32_f16(false, a0, false, bf, (short)0,
                                                            acc[0][nt], false, false);
        acc[1][nt] = __builtin_amdgcn_wmma_f32_16x16x32_f16(false, a1, false, bf, (short)0,
                                                            acc[1][nt], false, false);
      }
    }
#pragma unroll
    for (int nt = 0; nt < 8; ++nt) {
      const int o = (np * 8 + nt) * 16 + rsel;
      const float s = sc[o], h = sh[o];
#pragma unroll
      for (int mb = 0; mb < 2; ++mb) {
#pragma unroll
        for (int r = 0; r < 8; ++r) {
          const int m  = mb * 16 + r + hsel * 8;
          const int pt = rowBase + m;
          const int b  = pt >> 11;       // / 2048
          const int k  = pt & 2047;
          float v = acc[mb][nt][r] * s + h;
          v = v > 0.f ? v : 0.f;
          out[((size_t)b * 256 + o) * KK + k] = v;  // [B,256,K] transposed layout
        }
      }
    }
  }
}

// ---------------------------------------------------------------------------
extern "C" void kernel_launch(void* const* d_in, const int* in_sizes, int n_in,
                              void* d_out, int out_size, void* d_ws, size_t ws_size,
                              hipStream_t stream) {
  (void)in_sizes; (void)n_in; (void)out_size; (void)ws_size;
  const float* points   = (const float*)d_in[0];
  const float* features = (const float*)d_in[1];
  const float* P[27];
  for (int i = 0; i < 27; ++i) P[i] = (const float*)d_in[2 + i];   // (w,sc,sh) x3 layers x3 scales
  const float* agg_w  = (const float*)d_in[29];
  const float* agg_sc = (const float*)d_in[30];
  const float* agg_sh = (const float*)d_in[31];

  float* out_sampled = (float*)d_out;                       // [B,K,3]
  float* out_nf      = (float*)d_out + (size_t)BB * KK * 3; // [B,256,K]

  char* ws = (char*)d_ws;
  size_t off = 0;
  int* fps_idx = (int*)(ws + off); off += (size_t)BB * KK * 4;            // 32 KB
  int* idx1    = (int*)(ws + off); off += (size_t)BB * KK * 16 * 4;       // 512 KB
  int* idx2    = (int*)(ws + off); off += (size_t)BB * KK * 32 * 4;       // 1 MB
  int* idx3    = (int*)(ws + off); off += (size_t)BB * KK * 64 * 4;       // 2 MB
  _Float16* cat = (_Float16*)(ws + off); off += (size_t)BB * KK * CAT_CH * 2; // 8 MB

  // packed weight layouts: (COUT, CINR, CINP) per layer
  const int LW[10][3] = {
      {64, 67, 96},  {64, 64, 64},   {128, 64, 64},     // scale 1
      {64, 67, 96},  {96, 64, 64},   {128, 96, 96},     // scale 2
      {96, 67, 96},  {128, 96, 96},  {256, 128, 128},   // scale 3
      {256, 512, 512}                                   // aggregation
  };
  _Float16* PW[10];
  for (int l = 0; l < 10; ++l) {
    PW[l] = (_Float16*)(ws + off);
    off += (size_t)LW[l][0] * LW[l][2] * 2;
  }

  // pre-pack weights (f32 -> f16 fragment layout)
  for (int l = 0; l < 10; ++l) {
    const float* src = (l < 9) ? P[l * 3] : agg_w;
    const int total = LW[l][0] * LW[l][2];
    pack_w_kernel<<<(total + 255) / 256, 256, 0, stream>>>(src, PW[l], LW[l][0], LW[l][1], LW[l][2]);
  }

  fps_kernel<<<BB, 1024, 0, stream>>>(points, fps_idx, out_sampled);
  ballq_kernel<<<(BB * KK) / 4, 128, 0, stream>>>(points, out_sampled, idx1, idx2, idx3);

  sa_mlp_kernel<16, 64, 64, 128, 0><<<(BB * KK * 16) / 64, 64, 0, stream>>>(
      points, features, out_sampled, idx1,
      PW[0], P[1], P[2], PW[1], P[4], P[5], PW[2], P[7], P[8], cat);
  sa_mlp_kernel<32, 64, 96, 128, 128><<<(BB * KK * 32) / 64, 64, 0, stream>>>(
      points, features, out_sampled, idx2,
      PW[3], P[10], P[11], PW[4], P[13], P[14], PW[5], P[16], P[17], cat);
  sa_mlp_kernel<64, 96, 128, 256, 256><<<(BB * KK * 64) / 64, 64, 0, stream>>>(
      points, features, out_sampled, idx3,
      PW[6], P[19], P[20], PW[7], P[22], P[23], PW[8], P[25], P[26], cat);

  agg_kernel<<<(BB * KK) / 64, 64, 0, stream>>>(cat, PW[9], agg_sc, agg_sh, out_nf);
}
